// CombinedMSESSIMLoss_49246095015971
// MI455X (gfx1250) — matrix operand order repaired and checked by reading, not compiled, not run on hardware
//
#include <hip/hip_runtime.h>
#include <hip/hip_bf16.h>

typedef __attribute__((ext_vector_type(16))) _Float16 v16h;
typedef __attribute__((ext_vector_type(8)))  float    v8f;

#define IMG 512
#define OUT 502            // IMG - 11 + 1
#define PLANES 48          // 16 * 3
#define TILES_X 32         // ceil(502/16)
#define TOTAL_TILES (PLANES * TILES_X * TILES_X)   // 49152
#define WAVES_PER_BLOCK 8
#define SSIM_BLOCKS (TOTAL_TILES / WAVES_PER_BLOCK) // 6144

#define C1V 1.0e-4f
#define C2V 9.0e-4f

// ---------------------------------------------------------------------------
// Init: zero accumulators and precompute the per-lane banded-Gaussian WMMA
// operand tables (Bh in B-layout 32x16, Wv in A-layout 16x32) so the hot
// kernel fetches them with two b128 loads instead of ~130 VALU cndmasks.
// ---------------------------------------------------------------------------
__global__ void init_kernel(double* acc, v16h* bh_tab, v16h* wv_tab) {
    const int lane = threadIdx.x;   // 32 threads
    if (lane == 0) { acc[0] = 0.0; acc[1] = 0.0; }

    float g[11];
    float gs = 0.f;
#pragma unroll
    for (int k = 0; k < 11; ++k) {
        float d = (float)(k - 5);
        g[k] = __expf(-d * d * (1.0f / (2.0f * 1.5f * 1.5f)));
        gs += g[k];
    }
    float gi = 1.0f / gs;
#pragma unroll
    for (int k = 0; k < 11; ++k) g[k] *= gi;

    const int n  = lane & 15;
    const int hl = lane >> 4;

    v16h bh;   // B-layout 32x16: Bh[k][j] = g[k-j]
#pragma unroll
    for (int e = 0; e < 16; ++e) {
        int k = e + 8 * ((e >= 8 ? 1 : 0) + hl);
        int d = k - n;
        bh[e] = (d >= 0 && d < 11) ? (_Float16)g[d] : (_Float16)0.f;
    }
    v16h wv;   // A-layout 16x32: Wv[i][k] = g[k-i]
#pragma unroll
    for (int e = 0; e < 16; ++e) {
        int k = e + 16 * hl;
        int d = k - n;
        wv[e] = (d >= 0 && d < 11) ? (_Float16)g[d] : (_Float16)0.f;
    }
    bh_tab[lane] = bh;
    wv_tab[lane] = wv;
}

// ---------------------------------------------------------------------------
// MSE: pure memory-bound pass (hits L2 since both tensors fit in 192MB).
// ---------------------------------------------------------------------------
__global__ void __launch_bounds__(256) mse_kernel(const float4* __restrict__ x,
                                                  const float4* __restrict__ y,
                                                  double* __restrict__ acc, int n4) {
    int tid = blockIdx.x * blockDim.x + threadIdx.x;
    int stride = gridDim.x * blockDim.x;
    float s = 0.f;
    for (int i = tid; i < n4; i += stride) {
        float4 a = x[i], b = y[i];
        float d0 = a.x - b.x, d1 = a.y - b.y, d2 = a.z - b.z, d3 = a.w - b.w;
        s += d0 * d0 + d1 * d1 + d2 * d2 + d3 * d3;
    }
#pragma unroll
    for (int off = 16; off; off >>= 1) s += __shfl_xor(s, off, 32);
    if ((threadIdx.x & 31) == 0) atomicAdd(acc, (double)s);
}

// ---------------------------------------------------------------------------
// Helpers for the SSIM tile kernel.
// ---------------------------------------------------------------------------
__device__ __forceinline__ v16h load_row_f16_fast(const float* rp) {
    // rp points at (row, tc0 + 8*hl); elements e=0..7 -> rp[e], e=8..15 -> rp[16+e-8]
    const float4* p = (const float4*)rp;
    float4 q0 = p[0], q1 = p[1], q2 = p[4], q3 = p[5];
    v16h r;
    r[0]  = (_Float16)q0.x; r[1]  = (_Float16)q0.y; r[2]  = (_Float16)q0.z; r[3]  = (_Float16)q0.w;
    r[4]  = (_Float16)q1.x; r[5]  = (_Float16)q1.y; r[6]  = (_Float16)q1.z; r[7]  = (_Float16)q1.w;
    r[8]  = (_Float16)q2.x; r[9]  = (_Float16)q2.y; r[10] = (_Float16)q2.z; r[11] = (_Float16)q2.w;
    r[12] = (_Float16)q3.x; r[13] = (_Float16)q3.y; r[14] = (_Float16)q3.z; r[15] = (_Float16)q3.w;
    return r;
}

__device__ __forceinline__ v16h load_row_f16_clamped(const float* rowbase, int tc0, int hl) {
    v16h r;
#pragma unroll
    for (int e = 0; e < 16; ++e) {
        int k = e + 8 * ((e >= 8 ? 1 : 0) + hl);
        int c = tc0 + k; if (c > IMG - 1) c = IMG - 1;
        r[e] = (_Float16)rowbase[c];
    }
    return r;
}

__device__ __forceinline__ v16h mul16(v16h a, v16h b) {
    v16h r;
#pragma unroll
    for (int e = 0; e < 16; ++e) r[e] = a[e] * b[e];   // v_pk_mul_f16
    return r;
}

// ---------------------------------------------------------------------------
// SSIM: one wave32 per 16x16 output tile. Separable 11-tap Gaussian done as
// f16 WMMA GEMMs (K=26 padded into one 16x16x32 WMMA). 5 quantities
// (x, y, xx, yy, xy) * (2 horizontal + 1 vertical) = 15 WMMAs per tile.
// ---------------------------------------------------------------------------
__global__ void __launch_bounds__(256) ssim_kernel(const float* __restrict__ xg,
                                                   const float* __restrict__ yg,
                                                   double* __restrict__ acc,
                                                   const v16h* __restrict__ bh_tab,
                                                   const v16h* __restrict__ wv_tab) {
    __shared__ _Float16 lbuf_all[WAVES_PER_BLOCK][512];  // 32x16 f16 per wave

    const int lane = threadIdx.x & 31;
    const int wave = threadIdx.x >> 5;
    _Float16* lbuf = lbuf_all[wave];

    const int tile  = blockIdx.x * WAVES_PER_BLOCK + wave;
    const int plane = tile >> 10;          // 1024 tiles per plane
    const int t     = tile & 1023;
    const int ty    = t >> 5, tx = t & 31;
    const int tr0   = ty * 16;             // output-tile row origin
    const int tc0   = tx * 16;             // output-tile col origin

    const float* xp = xg + (size_t)plane * IMG * IMG;
    const float* yp = yg + (size_t)plane * IMG * IMG;

    const int n  = lane & 15;   // matrix row / output col index
    const int hl = lane >> 4;   // lane-half

    const v16h bh = bh_tab[lane];   // banded Gaussian, B-layout
    const v16h wv = wv_tab[lane];   // banded Gaussian, A-layout

    // --- Load the input region directly in A-layout order, convert to f16 ---
    v16h ax1, ax2, ay1, ay2;
    if (ty < 31 && tx < 31) {
        // interior fast path: aligned float4 loads, no clamping
        const float* x1 = xp + (tr0 + n) * IMG + tc0 + 8 * hl;
        const float* x2 = x1 + 16 * IMG;
        const float* y1 = yp + (tr0 + n) * IMG + tc0 + 8 * hl;
        const float* y2 = y1 + 16 * IMG;
        ax1 = load_row_f16_fast(x1);
        ax2 = load_row_f16_fast(x2);
        ay1 = load_row_f16_fast(y1);
        ay2 = load_row_f16_fast(y2);
    } else {
        // edge path: clamp rows/cols; clamped values only feed zero-weight K
        int r1 = tr0 + n;       if (r1 > IMG - 1) r1 = IMG - 1;
        int r2 = tr0 + 16 + n;  if (r2 > IMG - 1) r2 = IMG - 1;
        ax1 = load_row_f16_clamped(xp + r1 * IMG, tc0, hl);
        ax2 = load_row_f16_clamped(xp + r2 * IMG, tc0, hl);
        ay1 = load_row_f16_clamped(yp + r1 * IMG, tc0, hl);
        ay2 = load_row_f16_clamped(yp + r2 * IMG, tc0, hl);
    }

    // --- one separable Gaussian filter of a 16x16 tile: 3 WMMAs ------------
    auto conv = [&](v16h a1, v16h a2) -> v8f {
        v8f z = {0.f, 0.f, 0.f, 0.f, 0.f, 0.f, 0.f, 0.f};
        // horizontal pass: rows 0..15 and rows 16..25(+zero-weighted pad)
        v8f t1 = __builtin_amdgcn_wmma_f32_16x16x32_f16(false, a1, false, bh,
                                                        (short)0, z, false, false);
        v8f t2 = __builtin_amdgcn_wmma_f32_16x16x32_f16(false, a2, false, bh,
                                                        (short)0, z, false, false);
        // D-layout -> B-layout via LDS (f16). dst halfword = (n + 16*(k/16))*16 + k%16
        _Float16* p1 = lbuf + n * 16 + 8 * hl;          // t1: k = g + 8*hl
        _Float16* p2 = lbuf + (n + 16) * 16 + 8 * hl;   // t2: k = 16 + g + 8*hl
#pragma unroll
        for (int q = 0; q < 8; ++q) {
            p1[q] = (_Float16)t1[q];
            p2[q] = (_Float16)t2[q];
        }
        // B-layout read: lane reads 16 contiguous halfwords
        v16h b;
        const _Float16* rp = lbuf + lane * 16;
#pragma unroll
        for (int e = 0; e < 16; ++e) b[e] = rp[e];
        // vertical pass
        return __builtin_amdgcn_wmma_f32_16x16x32_f16(false, wv, false, b,
                                                      (short)0, z, false, false);
    };

    v8f mu_x = conv(ax1, ax2);
    v8f mu_y = conv(ay1, ay2);
    v8f sxx  = conv(mul16(ax1, ax1), mul16(ax2, ax2));
    v8f syy  = conv(mul16(ay1, ay1), mul16(ay2, ay2));
    v8f sxy  = conv(mul16(ax1, ay1), mul16(ax2, ay2));

    // --- SSIM map + masked accumulation (D-layout: M = g + 8*hl, N = n) ----
    float lsum = 0.f;
#pragma unroll
    for (int gg = 0; gg < 8; ++gg) {
        int orow = tr0 + gg + 8 * hl;
        int ocol = tc0 + n;
        if (orow < OUT && ocol < OUT) {
            float mx = mu_x[gg], my = mu_y[gg];
            float mx2 = mx * mx, my2 = my * my, mxy = mx * my;
            float vx  = sxx[gg] - mx2;
            float vy  = syy[gg] - my2;
            float vxy = sxy[gg] - mxy;
            float num = (2.f * mxy + C1V) * (2.f * vxy + C2V);
            float den = (mx2 + my2 + C1V) * (vx + vy + C2V);
            lsum += num / den;
        }
    }
#pragma unroll
    for (int off = 16; off; off >>= 1) lsum += __shfl_xor(lsum, off, 32);
    if (lane == 0) atomicAdd(acc, (double)lsum);
}

__global__ void finalize_kernel(const double* __restrict__ acc, float* __restrict__ out) {
    double ssim = acc[0] / (double)((long long)PLANES * OUT * OUT);
    double mse  = acc[1] / (double)((long long)PLANES * IMG * IMG);
    out[0] = (float)(0.8 * mse + 0.2 * (1.0 - ssim));
}

extern "C" void kernel_launch(void* const* d_in, const int* in_sizes, int n_in,
                              void* d_out, int out_size, void* d_ws, size_t ws_size,
                              hipStream_t stream) {
    (void)in_sizes; (void)n_in; (void)out_size; (void)ws_size;
    const float* pred   = (const float*)d_in[0];
    const float* target = (const float*)d_in[1];

    // ws layout: [0,16)   two double accumulators (ssim, mse)
    //            [64,1088)   bh table: 32 lanes x 32B
    //            [1088,2112) wv table: 32 lanes x 32B
    double* acc    = (double*)d_ws;
    v16h*   bh_tab = (v16h*)((char*)d_ws + 64);
    v16h*   wv_tab = (v16h*)((char*)d_ws + 64 + 32 * sizeof(v16h));
    float*  out    = (float*)d_out;

    init_kernel<<<1, 32, 0, stream>>>(acc, bh_tab, wv_tab);

    const int n4 = (PLANES * IMG * IMG) / 4;   // 3,145,728 float4s
    mse_kernel<<<1024, 256, 0, stream>>>((const float4*)pred, (const float4*)target,
                                         acc + 1, n4);

    ssim_kernel<<<SSIM_BLOCKS, 256, 0, stream>>>(pred, target, acc, bh_tab, wv_tab);

    finalize_kernel<<<1, 1, 0, stream>>>(acc, out);
}
